// GRUGNN_59210419143209
// MI455X (gfx1250) — matrix operand order repaired and checked by reading, not compiled, not run on hardware
//
#include <hip/hip_runtime.h>

typedef __attribute__((ext_vector_type(16))) __bf16 v16bf;
typedef __attribute__((ext_vector_type(8)))  __bf16 v8bf;
typedef __attribute__((ext_vector_type(8)))  float  v8f;

#define HDIM 128
#define W_ELEMS (3 * HDIM * HDIM)
#define LDS_STRIDE 136   // bf16 per weight row: 128 data + 8 pad (272 B) -> conflict-free ds_load_b128
#define BLOCK_THREADS 512
#define WAVES_PER_BLOCK (BLOCK_THREADS / 32)

// ---------------- Pass 0: zero the reduction mailbox ----------------
__global__ __launch_bounds__(256) void zero_kernel(float4* __restrict__ p, long long n4) {
    long long i = (long long)blockIdx.x * blockDim.x + threadIdx.x;
    if (i < n4) p[i] = make_float4(0.f, 0.f, 0.f, 0.f);
}

// ---------------- Pass 1: generic segment-sum via scatter-add ----------------
__global__ __launch_bounds__(256) void scatter_kernel(const float* __restrict__ h,
                                                      const int* __restrict__ src,
                                                      const int* __restrict__ dst,
                                                      float* __restrict__ red, int n) {
    long long i = (long long)blockIdx.x * blockDim.x + threadIdx.x;
    long long total = (long long)n * HDIM;
    if (i < total) {
        int e = (int)(i >> 7);
        int f = (int)(i & (HDIM - 1));
        atomicAdd(&red[(long long)dst[e] * HDIM + f], h[(long long)src[e] * HDIM + f]);
    }
}

// ---------------- Pass 2: fused dual-GEMM + GRU gating, bf16 WMMA ----------------

// B tile (32x16, K x N) per CDNA5 16-bit B layout, served from LDS:
// lane L holds output-column (L&15), K = (L>>4)*8 .. +7 and 16+(L>>4)*8 .. +7 -> two ds_load_b128.
__device__ inline v16bf load_b_lds(const __bf16* w, int colbase, int k0, int lane) {
    const __bf16* p = w + (colbase + (lane & 15)) * LDS_STRIDE + k0 + ((lane >> 4) << 3);
    v8bf lo = *(const v8bf*)p;
    v8bf hi = *(const v8bf*)(p + 16);
    return __builtin_shufflevector(lo, hi, 0,1,2,3,4,5,6,7,8,9,10,11,12,13,14,15);
}

// Convert 8 contiguous fp32 to bf16 into half of a v16bf A operand.
__device__ inline void cvt_a_chunk(const float* __restrict__ p, v16bf& v, int o) {
    float4 a = *(const float4*)p;
    float4 b = *(const float4*)(p + 4);
    v[o + 0] = (__bf16)a.x; v[o + 1] = (__bf16)a.y; v[o + 2] = (__bf16)a.z; v[o + 3] = (__bf16)a.w;
    v[o + 4] = (__bf16)b.x; v[o + 5] = (__bf16)b.y; v[o + 6] = (__bf16)b.z; v[o + 7] = (__bf16)b.w;
}

#define WMMA_BF16(A, B, C) \
    __builtin_amdgcn_wmma_f32_16x16x32_bf16(false, (A), false, (B), (short)0, (C), false, false)

__global__ __launch_bounds__(BLOCK_THREADS) void gru_wmma_kernel(
    const float* __restrict__ x, const float* __restrict__ red,
    const float* __restrict__ wih_g, const float* __restrict__ whh_g,
    const float* __restrict__ bih, const float* __restrict__ bhh,
    float* __restrict__ out, int n)
{
    // 204 KB of LDS: both weight matrices in bf16, padded rows (WGP has 320 KB).
    // One block per WGP; 16 waves share the staged weights (4 waves per SIMD32).
    __shared__ __bf16 wlds[2 * 3 * HDIM * LDS_STRIDE];

    // ---- cooperative staging: fp32 weights (global) -> bf16 (LDS) ----
    for (int m = 0; m < 2; ++m) {
        const float* wsrc = m ? whh_g : wih_g;
        __bf16* wdst = wlds + m * 3 * HDIM * LDS_STRIDE;
        for (int c = threadIdx.x; c < W_ELEMS / 4; c += BLOCK_THREADS) {
            int idx = c << 2;              // 4 consecutive elements
            int row = idx >> 7;            // / HDIM
            int col = idx & (HDIM - 1);
            float4 v = *(const float4*)(wsrc + idx);
            __bf16* d = wdst + row * LDS_STRIDE + col;
            d[0] = (__bf16)v.x; d[1] = (__bf16)v.y; d[2] = (__bf16)v.z; d[3] = (__bf16)v.w;
        }
    }
    __syncthreads();

    const __bf16* wih = wlds;
    const __bf16* whh = wlds + 3 * HDIM * LDS_STRIDE;

    const int lane = threadIdx.x & 31;
    const int wave = threadIdx.x >> 5;
    const long long base = ((long long)blockIdx.x * WAVES_PER_BLOCK + wave) * 32;  // 32 nodes/wave
    if (base >= n) return;  // wave-uniform (after barrier): EXEC all-1s for WMMA

    const int row16 = lane & 15;
    const int half  = lane >> 4;

    // A tiles (16x32 bf16) for x and red: 2 M-subtiles x 4 K-steps each.
    // A layout: lane<16 holds K = ks*32 + {0..7, 16..23}; lane>=16 holds {8..15, 24..31}.
    v16bf ax[2][4], ar[2][4];
#pragma unroll
    for (int s = 0; s < 2; ++s) {
        long long arow = base + s * 16 + row16;
        if (arow >= n) arow = n - 1;   // clamp for tail; stores masked below
        const float* xr = x   + arow * HDIM;
        const float* rr = red + arow * HDIM;
#pragma unroll
        for (int ks = 0; ks < 4; ++ks) {
            int k0 = ks * 32 + half * 8;
            int k1 = ks * 32 + 16 + half * 8;
            cvt_a_chunk(xr + k0, ax[s][ks], 0);
            cvt_a_chunk(xr + k1, ax[s][ks], 8);
            cvt_a_chunk(rr + k0, ar[s][ks], 0);
            cvt_a_chunk(rr + k1, ar[s][ks], 8);
        }
    }

    // 8 column tiles of 16 covering HDIM=128 output columns.
    // Gate fusion: r/z gates only need (i_r + h_r), (i_z + h_z) -> accumulate both
    // GEMMs into one accumulator per gate. Only the n-gate stays split.
    for (int ct = 0; ct < 8; ++ct) {
        v8f c_r[2] = {}, c_z[2] = {}, c_in[2] = {}, c_hn[2] = {};  // 8 accumulators
#pragma unroll
        for (int ks = 0; ks < 4; ++ks) {
            int k0 = ks * 32;
            v16bf b;
            b = load_b_lds(wih, ct * 16, k0, lane);             // W_ir
            c_r[0] = WMMA_BF16(ax[0][ks], b, c_r[0]);
            c_r[1] = WMMA_BF16(ax[1][ks], b, c_r[1]);
            b = load_b_lds(whh, ct * 16, k0, lane);             // W_hr (chained into c_r)
            c_r[0] = WMMA_BF16(ar[0][ks], b, c_r[0]);
            c_r[1] = WMMA_BF16(ar[1][ks], b, c_r[1]);
            b = load_b_lds(wih, HDIM + ct * 16, k0, lane);      // W_iz
            c_z[0] = WMMA_BF16(ax[0][ks], b, c_z[0]);
            c_z[1] = WMMA_BF16(ax[1][ks], b, c_z[1]);
            b = load_b_lds(whh, HDIM + ct * 16, k0, lane);      // W_hz (chained into c_z)
            c_z[0] = WMMA_BF16(ar[0][ks], b, c_z[0]);
            c_z[1] = WMMA_BF16(ar[1][ks], b, c_z[1]);
            b = load_b_lds(wih, 2 * HDIM + ct * 16, k0, lane);  // W_in
            c_in[0] = WMMA_BF16(ax[0][ks], b, c_in[0]);
            c_in[1] = WMMA_BF16(ax[1][ks], b, c_in[1]);
            b = load_b_lds(whh, 2 * HDIM + ct * 16, k0, lane);  // W_hn (separate acc)
            c_hn[0] = WMMA_BF16(ar[0][ks], b, c_hn[0]);
            c_hn[1] = WMMA_BF16(ar[1][ks], b, c_hn[1]);
        }

        // Epilogue: C/D layout -> row = base + s*16 + r + 8*half, col = ct*16 + (lane&15)
        int col = ct * 16 + row16;
        float b_r  = bih[col] + bhh[col];
        float b_z  = bih[HDIM + col] + bhh[HDIM + col];
        float b_in = bih[2 * HDIM + col];
        float b_hn = bhh[2 * HDIM + col];
#pragma unroll
        for (int s = 0; s < 2; ++s) {
#pragma unroll
            for (int r = 0; r < 8; ++r) {
                long long row = base + s * 16 + r + half * 8;
                if (row < n) {
                    float rg = __builtin_amdgcn_rcpf(1.0f + __expf(-(c_r[s][r] + b_r)));
                    float zg = __builtin_amdgcn_rcpf(1.0f + __expf(-(c_z[s][r] + b_z)));
                    float t  = (c_in[s][r] + b_in) + rg * (c_hn[s][r] + b_hn);
                    float e2 = __expf(-2.0f * t);           // tanh(t) = (1-e2)/(1+e2)
                    float ng = (1.0f - e2) * __builtin_amdgcn_rcpf(1.0f + e2);
                    float rv = red[row * HDIM + col];
                    out[row * HDIM + col] = (1.0f - zg) * ng + zg * rv;
                }
            }
        }
    }
}

extern "C" void kernel_launch(void* const* d_in, const int* in_sizes, int n_in,
                              void* d_out, int out_size, void* d_ws, size_t ws_size,
                              hipStream_t stream) {
    const float* x    = (const float*)d_in[0];
    const float* h    = (const float*)d_in[1];
    const float* w_ih = (const float*)d_in[2];
    const float* w_hh = (const float*)d_in[3];
    const float* b_ih = (const float*)d_in[4];
    const float* b_hh = (const float*)d_in[5];
    const int*   src  = (const int*)d_in[6];
    const int*   dst  = (const int*)d_in[7];
    const int n = in_sizes[6];  // src has N elements

    // Workspace: [red: N*HDIM f32]
    float* red = (float*)d_ws;

    long long n4 = (long long)n * HDIM / 4;
    zero_kernel<<<(int)((n4 + 255) / 256), 256, 0, stream>>>((float4*)red, n4);

    long long tot = (long long)n * HDIM;
    scatter_kernel<<<(int)((tot + 255) / 256), 256, 0, stream>>>(h, src, dst, red, n);

    int tiles  = (n + 31) / 32;                                  // 32 nodes per wave
    int blocks = (tiles + WAVES_PER_BLOCK - 1) / WAVES_PER_BLOCK; // 16 waves per block
    gru_wmma_kernel<<<blocks, BLOCK_THREADS, 0, stream>>>(x, red, w_ih, w_hh, b_ih, b_hh,
                                                          (float*)d_out, n);
}
